// MultiHeadAttention_40793599377674
// MI455X (gfx1250) — matrix operand order repaired and checked by reading, not compiled, not run on hardware
//
#include <hip/hip_runtime.h>

typedef __attribute__((ext_vector_type(16))) _Float16 v16h;
typedef __attribute__((ext_vector_type(8)))  _Float16 v8h;
typedef __attribute__((ext_vector_type(8)))  float    v8f;

#define EMBED 512
#define SEQ   2048
#define BATCH 8
// 1/sqrt(512): both the EqualizedLinear scale and the attention scale
#define INV_SQRT_512 0.04419417382415922f

static __device__ __forceinline__ v8f wmma16(v16h a, v16h b, v8f c) {
    // 8 args: (neg_a, A, neg_b, B, c_mod, C, reuse_a, reuse_b)
    return __builtin_amdgcn_wmma_f32_16x16x32_f16(false, a, false, b, (short)0, c,
                                                  false, false);
}

// ---- A-matrix fragment: 16x32 (MxK), source row-major by M ----
// Lane (m = lane&15, h = lane>>4) holds K = k0+16h .. +15 contiguously.
static __device__ __forceinline__ v16h load_A_f32(const float* __restrict__ base,
                                                  int ld, int row, int k0, int h) {
    const float* p = base + (size_t)row * ld + k0 + (h << 4);
    v16h a;
    #pragma unroll
    for (int j = 0; j < 8; ++j) {
        a[j]     = (_Float16)p[j];
        a[j + 8] = (_Float16)p[j + 8];
    }
    return a;
}

static __device__ __forceinline__ v16h load_A_f16(const _Float16* __restrict__ base,
                                                  int ld, int row, int k0, int h) {
    const _Float16* p = base + (size_t)row * ld + k0 + (h << 4);
    v8h r0 = *(const v8h*)(p);
    v8h r1 = *(const v8h*)(p + 8);
    v16h a;
    #pragma unroll
    for (int j = 0; j < 8; ++j) { a[j] = r0[j]; a[j + 8] = r1[j]; }
    return a;
}

// ---- B-matrix fragment: 32x16 (KxN) ----
// Lane (n = lane&15, h = lane>>4) holds K = k0+16h .. +15 contiguously.
// Memory matrix is mem[n][k] row-major -> one 32-byte contiguous load.
static __device__ __forceinline__ v16h load_B_rowk_f16(const _Float16* __restrict__ base,
                                                       int ld, int col, int k0, int h) {
    const _Float16* p = base + (size_t)col * ld + k0 + (h << 4);
    return *(const v16h*)p;   // 32 bytes, 32B-aligned
}

static __device__ __forceinline__ v16h load_B_rowk_f32(const float* __restrict__ base,
                                                       int ld, int col, int k0, int h) {
    const float* p = base + (size_t)col * ld + k0 + (h << 4);
    v16h b;
    #pragma unroll
    for (int j = 0; j < 16; ++j) b[j] = (_Float16)p[j];
    return b;
}

// =====================================================================
// Stage 1: Y = (X @ W^T) * 1/sqrt(512) in f16.
// Macro-tile 32(rows) x 64(e) per wave: 2 A-frags x 4 B-frags, 8 WMMA/K-step.
// TRANS=false: Y[row][e] row-major (for qp / kp).
// TRANS=true : Y stored [b][e][m] (for vp, so stage 4's B is contiguous in K);
//              packed 8xf16 b128 stores.
// =====================================================================
template <bool TRANS>
__global__ __launch_bounds__(256) void proj_kernel(const float* __restrict__ X,
                                                   const float* __restrict__ W,
                                                   _Float16* __restrict__ Y) {
    int wave = blockIdx.x * 8 + (threadIdx.x >> 5);   // 0..4095
    int lane = threadIdx.x & 31;
    int tm = wave >> 3;       // 0..511  (32-row macro tile)
    int tn = wave & 7;        // 0..7    (64-col macro tile, e)
    int h = lane >> 4, l = lane & 15;
    int row0 = tm * 32, e0 = tn * 64;

    v8f acc[2][4] = {};
    for (int k0 = 0; k0 < EMBED; k0 += 32) {
        v16h a[2], b[4];
        #pragma unroll
        for (int i = 0; i < 2; ++i)
            a[i] = load_A_f32(X, EMBED, row0 + i * 16 + l, k0, h);
        #pragma unroll
        for (int j = 0; j < 4; ++j)
            b[j] = load_B_rowk_f32(W, EMBED, e0 + j * 16 + l, k0, h);
        #pragma unroll
        for (int i = 0; i < 2; ++i)
            #pragma unroll
            for (int j = 0; j < 4; ++j)
                acc[i][j] = wmma16(a[i], b[j], acc[i][j]);
    }

    if (TRANS) {
        // Y layout: [b][e][m], b = row0>>11, m = row0&2047 (tiles never cross b)
        _Float16* Yb = Y + (size_t)(row0 >> 11) * EMBED * SEQ;
        int mbase = (row0 & (SEQ - 1)) + 8 * h;
        #pragma unroll
        for (int i = 0; i < 2; ++i)
            #pragma unroll
            for (int j = 0; j < 4; ++j) {
                v8h pk;
                #pragma unroll
                for (int c = 0; c < 8; ++c)
                    pk[c] = (_Float16)(acc[i][j][c] * INV_SQRT_512);
                *(v8h*)(Yb + (size_t)(e0 + j * 16 + l) * SEQ + mbase + i * 16) = pk;
            }
    } else {
        #pragma unroll
        for (int i = 0; i < 2; ++i)
            #pragma unroll
            for (int j = 0; j < 4; ++j)
                #pragma unroll
                for (int c = 0; c < 8; ++c)
                    Y[(size_t)(row0 + i * 16 + c + 8 * h) * EMBED + e0 + j * 16 + l] =
                        (_Float16)(acc[i][j][c] * INV_SQRT_512);
    }
}

// =====================================================================
// Stage 2: S[b][n][m] = (qp . kp) * 1/sqrt(512), f32 into d_out attn region.
// Macro-tile 32(n) x 64(m) per wave; per batch 64x32 = 2048 waves.
// =====================================================================
__global__ __launch_bounds__(256) void scores_kernel(const _Float16* __restrict__ QP,
                                                     const _Float16* __restrict__ KP,
                                                     float* __restrict__ S) {
    int b = blockIdx.y;
    int wave = blockIdx.x * 8 + (threadIdx.x >> 5);   // 0..2047
    int lane = threadIdx.x & 31;
    int tm = wave >> 5;    // 0..63 (n macro tile)
    int tn = wave & 31;    // 0..31 (m macro tile)
    int h = lane >> 4, l = lane & 15;
    int n0 = tm * 32, m0 = tn * 64;

    const _Float16* qp = QP + (size_t)b * SEQ * EMBED;
    const _Float16* kp = KP + (size_t)b * SEQ * EMBED;

    v8f acc[2][4] = {};
    for (int k0 = 0; k0 < EMBED; k0 += 32) {
        v16h a[2], bm[4];
        #pragma unroll
        for (int i = 0; i < 2; ++i)
            a[i] = load_A_f16(qp, EMBED, n0 + i * 16 + l, k0, h);
        #pragma unroll
        for (int j = 0; j < 4; ++j)
            bm[j] = load_B_rowk_f16(kp, EMBED, m0 + j * 16 + l, k0, h);
        #pragma unroll
        for (int i = 0; i < 2; ++i)
            #pragma unroll
            for (int j = 0; j < 4; ++j)
                acc[i][j] = wmma16(a[i], bm[j], acc[i][j]);
    }

    float* Sb = S + (size_t)b * SEQ * SEQ;
    #pragma unroll
    for (int i = 0; i < 2; ++i)
        #pragma unroll
        for (int j = 0; j < 4; ++j)
            #pragma unroll
            for (int c = 0; c < 8; ++c)
                Sb[(size_t)(n0 + i * 16 + c + 8 * h) * SEQ + m0 + j * 16 + l] =
                    acc[i][j][c] * INV_SQRT_512;
}

// =====================================================================
// Stage 3: row-wise softmax in place. One 256-thread block per row of 2048.
// =====================================================================
__global__ __launch_bounds__(256) void softmax_kernel(float* __restrict__ A) {
    __shared__ float sdata[256];
    int tid = threadIdx.x;
    float* p = A + (size_t)blockIdx.x * SEQ;

    float v[8];
    float lmax = -3.402823466e38f;
    #pragma unroll
    for (int i = 0; i < 8; ++i) { v[i] = p[tid + 256 * i]; lmax = fmaxf(lmax, v[i]); }

    sdata[tid] = lmax; __syncthreads();
    for (int s = 128; s > 0; s >>= 1) {
        if (tid < s) sdata[tid] = fmaxf(sdata[tid], sdata[tid + s]);
        __syncthreads();
    }
    float rmax = sdata[0];
    __syncthreads();

    float lsum = 0.f;
    #pragma unroll
    for (int i = 0; i < 8; ++i) { v[i] = __expf(v[i] - rmax); lsum += v[i]; }

    sdata[tid] = lsum; __syncthreads();
    for (int s = 128; s > 0; s >>= 1) {
        if (tid < s) sdata[tid] += sdata[tid + s];
        __syncthreads();
    }
    float inv = 1.0f / sdata[0];

    #pragma unroll
    for (int i = 0; i < 8; ++i) p[tid + 256 * i] = v[i] * inv;
}

// =====================================================================
// Stage 4: out[b][n][e] = attn[b][n][:] . vpT[b][e][:], f32 out.
// attn (f32, [0,1]) converted to f16 in-register; vpT gives contiguous B frags.
// Macro-tile 32(n) x 64(e); per batch 64x8 = 512 waves; K = 2048 -> 64 steps.
// =====================================================================
__global__ __launch_bounds__(256) void out_kernel(const float* __restrict__ ATT,
                                                  const _Float16* __restrict__ VPT,
                                                  float* __restrict__ O) {
    int b = blockIdx.y;
    int wave = blockIdx.x * 8 + (threadIdx.x >> 5);   // 0..511
    int lane = threadIdx.x & 31;
    int tm = wave >> 3;   // 0..63 (n macro tile)
    int tn = wave & 7;    // 0..7  (e macro tile)
    int h = lane >> 4, l = lane & 15;
    int n0 = tm * 32, e0 = tn * 64;

    const float*    attn = ATT + (size_t)b * SEQ * SEQ;
    const _Float16* vpt  = VPT + (size_t)b * EMBED * SEQ;   // [e][m]

    v8f acc[2][4] = {};
    for (int k0 = 0; k0 < SEQ; k0 += 32) {
        v16h a[2], bm[4];
        #pragma unroll
        for (int i = 0; i < 2; ++i)
            a[i] = load_A_f32(attn, SEQ, n0 + i * 16 + l, k0, h);
        #pragma unroll
        for (int j = 0; j < 4; ++j)
            bm[j] = load_B_rowk_f16(vpt, SEQ, e0 + j * 16 + l, k0, h);
        #pragma unroll
        for (int i = 0; i < 2; ++i)
            #pragma unroll
            for (int j = 0; j < 4; ++j)
                acc[i][j] = wmma16(a[i], bm[j], acc[i][j]);
    }

    float* Ob = O + (size_t)b * SEQ * EMBED;
    #pragma unroll
    for (int i = 0; i < 2; ++i)
        #pragma unroll
        for (int j = 0; j < 4; ++j)
            #pragma unroll
            for (int c = 0; c < 8; ++c)
                Ob[(size_t)(n0 + i * 16 + c + 8 * h) * EMBED + e0 + j * 16 + l] =
                    acc[i][j][c];
}

extern "C" void kernel_launch(void* const* d_in, const int* in_sizes, int n_in,
                              void* d_out, int out_size, void* d_ws, size_t ws_size,
                              hipStream_t stream) {
    const float* q  = (const float*)d_in[0];
    const float* k  = (const float*)d_in[1];
    const float* v  = (const float*)d_in[2];
    const float* Wq = (const float*)d_in[3];
    const float* Wk = (const float*)d_in[4];
    const float* Wv = (const float*)d_in[5];

    float* out  = (float*)d_out;                                  // [8,2048,512]
    float* attn = out + (size_t)BATCH * SEQ * EMBED;              // [8,2048,2048]

    const size_t projN = (size_t)BATCH * SEQ * EMBED;             // 8.4M elems
    _Float16* qp  = (_Float16*)d_ws;                              // [b][n][e] f16
    _Float16* kp  = qp + projN;                                   // [b][m][e] f16
    _Float16* vpt = kp + projN;                                   // [b][e][m] f16

    dim3 blk(256);
    // 16384x512 GEMM: 512*8 = 4096 macro-tiles, 8 waves/block -> 512 blocks
    proj_kernel<false><<<512, blk, 0, stream>>>(q, Wq, qp);
    proj_kernel<false><<<512, blk, 0, stream>>>(k, Wk, kp);
    proj_kernel<true ><<<512, blk, 0, stream>>>(v, Wv, vpt);
    // per batch: 64*32 = 2048 macro-tiles -> 256 blocks
    scores_kernel<<<dim3(256, BATCH), blk, 0, stream>>>(qp, kp, attn);
    // one block per attention row
    softmax_kernel<<<BATCH * SEQ, blk, 0, stream>>>(attn);
    // per batch: 64*8 = 512 macro-tiles -> 64 blocks
    out_kernel<<<dim3(64, BATCH), blk, 0, stream>>>(attn, vpt, out);
}
